// RecursiveCognitiveBlock_70128226009493
// MI455X (gfx1250) — matrix operand order repaired and checked by reading, not compiled, not run on hardware
//
#include <hip/hip_runtime.h>

// ---------------------------------------------------------------------------
// RecursiveCognitiveBlock for MI455X (gfx1250, wave32, WMMA + TDM)
//
// Shapes: B=2,S=1024 -> T=2048 tokens, D=1024, F=2048, E=8 experts, LOOPS=2.
// bf16 WMMA (v_wmma_f32_16x16x32_bf16) for both expert GEMMs, fp32 accumulate.
// Tiles are staged LDS-side by the Tensor Data Mover (tensor_load_to_lds,
// TENSORcnt-synchronized), double buffered, so the 8 compute waves spend their
// issue slots on WMMA instead of vmem staging.
// ---------------------------------------------------------------------------

typedef unsigned short u16;
typedef __attribute__((ext_vector_type(16))) __bf16 v16bf;
typedef __attribute__((ext_vector_type(8)))  __bf16 v8bf;
typedef __attribute__((ext_vector_type(8)))  float  v8f;
typedef unsigned int u32x4 __attribute__((ext_vector_type(4)));
typedef int          i32x4 __attribute__((ext_vector_type(4)));
typedef int          i32x8 __attribute__((ext_vector_type(8)));

#define T_TOK 2048
#define D_DIM 1024
#define F_DIM 2048
#define N_EXP 8
#define EF    (N_EXP * F_DIM)   // 16384
#define KC    64                // K elements staged per chunk (2 WMMA k-steps)

static __device__ __forceinline__ u16 f2bf(float f) {
  unsigned u = __float_as_uint(f);
  unsigned r = u + 0x7FFFu + ((u >> 16) & 1u);   // round-to-nearest-even
  return (u16)(r >> 16);
}
static __device__ __forceinline__ float bf2f(u16 h) {
  return __uint_as_float(((unsigned)h) << 16);
}

// ---------------------------------------------------------------------------
// Tensor Data Mover: 2D tile (rows x cols bf16) global -> LDS, packed
// row-major in LDS (tile_dim0 = cols elements contiguous per row).
// D# per CDNA5 ISA ch. 8; data_size=1 (2 bytes), type=2 ("image").
// ---------------------------------------------------------------------------
static __device__ __forceinline__ void tdm_load_tile_2d(unsigned lds_byte_addr,
                                                        const u16* gptr,
                                                        unsigned rows, unsigned cols,
                                                        unsigned row_stride_elems) {
  unsigned long long ga = (unsigned long long)gptr;
  u32x4 g0;
  g0[0] = 1u;                                            // count=1, user load desc
  g0[1] = lds_byte_addr;                                 // lds_addr
  g0[2] = (unsigned)(ga & 0xffffffffu);                  // global_addr[31:0]
  g0[3] = (unsigned)((ga >> 32) & 0x1ffffffu) | (2u << 30);  // [56:32] | type=2
  unsigned td0 = row_stride_elems;                       // tensor_dim0 (width)
  unsigned td1 = 0x40000000u;                            // tensor_dim1 (never OOB)
  i32x8 g1;
  g1[0] = (int)(1u << 16);                               // data_size=1 -> 2 bytes
  g1[1] = (int)((td0 & 0xffffu) << 16);                  // tensor_dim0[15:0]
  g1[2] = (int)((td0 >> 16) | ((td1 & 0xffffu) << 16));  // dim0[31:16] | dim1[15:0]
  g1[3] = (int)((td1 >> 16) | (cols << 16));             // dim1[31:16] | tile_dim0
  g1[4] = (int)rows;                                     // tile_dim1 | tile_dim2=0
  g1[5] = (int)row_stride_elems;                         // tensor_dim0_stride[31:0]
  g1[6] = 0;                                             // stride hi | dim1_stride lo
  g1[7] = 0;
  i32x4 gz;
  gz[0] = 0; gz[1] = 0; gz[2] = 0; gz[3] = 0;
#if __clang_major__ >= 23
  i32x8 gz8;
#pragma unroll
  for (int i = 0; i < 8; ++i) gz8[i] = 0;
  __builtin_amdgcn_tensor_load_to_lds(g0, g1, gz, gz, gz8, 0);
#else
  __builtin_amdgcn_tensor_load_to_lds(g0, g1, gz, gz, 0);
#endif
}

// A-fragment (16x32 bf16): lane = M (mod 16); lanes 0-15 hold K 0-7 & 16-23,
// lanes 16-31 hold K 8-15 & 24-31 (per CDNA5 ISA 7.12.2).
static __device__ __forceinline__ v16bf load_frag_a(const u16* rowptr, int lane) {
  int ho = (lane >> 4) * 8;  // 0 or 8
  v8bf lo = *(const v8bf*)(const void*)(rowptr + ho);
  v8bf hi = *(const v8bf*)(const void*)(rowptr + ho + 16);
  return __builtin_shufflevector(lo, hi, 0,1,2,3,4,5,6,7,8,9,10,11,12,13,14,15);
}
// B-fragment (32x16 bf16): lane = N (mod 16); lanes 0-15 hold K 0-7 & 16-23,
// lanes 16-31 hold K 8-15 & 24-31 (same K order as A).
static __device__ __forceinline__ v16bf load_frag_b(const u16* rowptr, int lane) {
  int kb = (lane >> 4) * 8;  // 0 or 8
  v8bf lo = *(const v8bf*)(const void*)(rowptr + kb);
  v8bf hi = *(const v8bf*)(const void*)(rowptr + kb + 16);
  return __builtin_shufflevector(lo, hi, 0,1,2,3,4,5,6,7,8,9,10,11,12,13,14,15);
}

static __device__ __forceinline__ v8f vzero8() {
  v8f z;
#pragma unroll
  for (int i = 0; i < 8; ++i) z[i] = 0.0f;
  return z;
}

// ---------------------------------------------------------------------------
// fp32 -> bf16 weight conversion
// ---------------------------------------------------------------------------
__global__ __launch_bounds__(256) void cvt_bf16_kernel(const float* __restrict__ in,
                                                       u16* __restrict__ out, int n) {
  int i = blockIdx.x * 256 + threadIdx.x;
  if (i < n) out[i] = f2bf(in[i]);
}

// ---------------------------------------------------------------------------
// RMSNorm: one block per token, 256 threads, 4 floats/thread.
// ---------------------------------------------------------------------------
__global__ __launch_bounds__(256) void rmsnorm_kernel(const float* __restrict__ z,
                                                      const float* __restrict__ nw,
                                                      u16* __restrict__ xn) {
  int t = blockIdx.x, tid = threadIdx.x;
  const float4 v = ((const float4*)(z + (long)t * D_DIM))[tid];
  float ssq = v.x * v.x + v.y * v.y + v.z * v.z + v.w * v.w;
#pragma unroll
  for (int o = 16; o; o >>= 1) ssq += __shfl_xor(ssq, o, 32);
  __shared__ float wsum[8];
  int lane = tid & 31, wid = tid >> 5;
  if (lane == 0) wsum[wid] = ssq;
  __syncthreads();
  if (tid == 0) {
    float tot = 0.f;
#pragma unroll
    for (int i = 0; i < 8; ++i) tot += wsum[i];
    wsum[0] = tot;
  }
  __syncthreads();
  float scale = rsqrtf(wsum[0] * (1.0f / (float)D_DIM) + 1e-6f);
  const float4 w = ((const float4*)nw)[tid];
  unsigned lo = (unsigned)f2bf(v.x * scale * w.x) | ((unsigned)f2bf(v.y * scale * w.y) << 16);
  unsigned hi = (unsigned)f2bf(v.z * scale * w.z) | ((unsigned)f2bf(v.w * scale * w.w) << 16);
  ((uint2*)(xn + (long)t * D_DIM))[tid] = make_uint2(lo, hi);
}

// ---------------------------------------------------------------------------
// Router: one block per token. 4 macro logits + 8 micro logits, softmax,
// top-2 groups (renormalized), top-1 expert per group -> combine[T,8].
// ---------------------------------------------------------------------------
__global__ __launch_bounds__(256) void router_kernel(const u16* __restrict__ xn,
                                                     const float* __restrict__ wmac,
                                                     const float* __restrict__ wmic,
                                                     float* __restrict__ combine) {
  int t = blockIdx.x, tid = threadIdx.x;
  float acc[12];
#pragma unroll
  for (int i = 0; i < 12; ++i) acc[i] = 0.f;
  const u16* xr = xn + (long)t * D_DIM;
  for (int d = tid; d < D_DIM; d += 256) {
    float x = bf2f(xr[d]);
#pragma unroll
    for (int g = 0; g < 4; ++g) acc[g] += x * wmac[g * D_DIM + d];
#pragma unroll
    for (int k = 0; k < 8; ++k) acc[4 + k] += x * wmic[k * D_DIM + d];
  }
  __shared__ float red[12][8];
  int lane = tid & 31, wid = tid >> 5;
#pragma unroll
  for (int i = 0; i < 12; ++i) {
    float v = acc[i];
#pragma unroll
    for (int o = 16; o; o >>= 1) v += __shfl_xor(v, o, 32);
    if (lane == 0) red[i][wid] = v;
  }
  __syncthreads();
  if (tid == 0) {
    float L[12];
#pragma unroll
    for (int i = 0; i < 12; ++i) {
      float s = 0.f;
#pragma unroll
      for (int w = 0; w < 8; ++w) s += red[i][w];
      L[i] = s;
    }
    float mx = fmaxf(fmaxf(L[0], L[1]), fmaxf(L[2], L[3]));
    float p[4], ps = 0.f;
#pragma unroll
    for (int g = 0; g < 4; ++g) { p[g] = __expf(L[g] - mx); ps += p[g]; }
#pragma unroll
    for (int g = 0; g < 4; ++g) p[g] /= ps;
    int g0 = 0;
#pragma unroll
    for (int g = 1; g < 4; ++g) if (p[g] > p[g0]) g0 = g;
    int g1 = (g0 == 0) ? 1 : 0;
#pragma unroll
    for (int g = 0; g < 4; ++g) if (g != g0 && p[g] > p[g1]) g1 = g;
    float denom = p[g0] + p[g1] + 1e-9f;
    float cw[8];
#pragma unroll
    for (int i = 0; i < 8; ++i) cw[i] = 0.f;
    int gs[2] = {g0, g1};
    float wgv[2] = {p[g0] / denom, p[g1] / denom};
#pragma unroll
    for (int s = 0; s < 2; ++s) {
      int g = gs[s];
      float a = L[4 + 2 * g], b = L[4 + 2 * g + 1];
      int j = (b > a) ? 1 : 0;
      float m = fmaxf(a, b);
      float ea = __expf(a - m), eb = __expf(b - m);
      float pj = ((j == 1) ? eb : ea) / (ea + eb);
      float lv = pj / (pj + 1e-9f);
      cw[g * 2 + j] += lv * wgv[s];
    }
#pragma unroll
    for (int i = 0; i < 8; ++i) combine[(long)t * 8 + i] = cw[i];
  }
}

// ---------------------------------------------------------------------------
// Up GEMM: H[t, n] = silu( xn[t,:] . w1bf[n,:] ) * combine[t, n/F]
// M=2048, N=16384, K=1024. 128x128 tile / WG, 8 waves of 32x64.
// A/B tiles (128 x KC bf16) staged by TDM, double buffered.
// ---------------------------------------------------------------------------
__global__ __launch_bounds__(256) void moe_up_kernel(const u16* __restrict__ xn,
                                                     const u16* __restrict__ w1bf,
                                                     const float* __restrict__ combine,
                                                     u16* __restrict__ H) {
  const int nbase = blockIdx.x * 128;
  const int mbase = blockIdx.y * 128;
  const int e = nbase >> 11;  // nbase / F_DIM
  const int tid = threadIdx.x;
  const int lane = tid & 31, wid = tid >> 5;
  const int wm = (wid & 3) * 32, wn = (wid >> 2) * 64;
  const int lr = lane & 15;

  __shared__ __align__(16) u16 As[2][128 * KC];
  __shared__ __align__(16) u16 Bs[2][128 * KC];
  __shared__ float cmb[128];

  if (tid < 128) cmb[tid] = combine[(long)(mbase + tid) * N_EXP + e];

  const u16* gA = xn + (long)mbase * D_DIM;    // tile row 0, k=0
  const u16* gB = w1bf + (long)nbase * D_DIM;
  unsigned ldsA[2] = {(unsigned)(unsigned long long)(void*)&As[0][0],
                      (unsigned)(unsigned long long)(void*)&As[1][0]};
  unsigned ldsB[2] = {(unsigned)(unsigned long long)(void*)&Bs[0][0],
                      (unsigned)(unsigned long long)(void*)&Bs[1][0]};

  if (wid == 0) {
    tdm_load_tile_2d(ldsA[0], gA, 128, KC, D_DIM);
    tdm_load_tile_2d(ldsB[0], gB, 128, KC, D_DIM);
  }
  __builtin_amdgcn_s_wait_tensorcnt(0);
  __syncthreads();

  v8f acc[2][4];
#pragma unroll
  for (int mi = 0; mi < 2; ++mi)
#pragma unroll
    for (int ni = 0; ni < 4; ++ni) acc[mi][ni] = vzero8();

  const int KT = D_DIM / KC;  // 16
  for (int kc = 0; kc < KT; ++kc) {
    const int cur = kc & 1, nxt = cur ^ 1;
    if (kc + 1 < KT && wid == 0) {
      tdm_load_tile_2d(ldsA[nxt], gA + (kc + 1) * KC, 128, KC, D_DIM);
      tdm_load_tile_2d(ldsB[nxt], gB + (kc + 1) * KC, 128, KC, D_DIM);
    }
#pragma unroll
    for (int ks = 0; ks < 2; ++ks) {
      v16bf af[2], bfr[4];
#pragma unroll
      for (int mi = 0; mi < 2; ++mi)
        af[mi] = load_frag_a(&As[cur][(wm + mi * 16 + lr) * KC + ks * 32], lane);
#pragma unroll
      for (int ni = 0; ni < 4; ++ni)
        bfr[ni] = load_frag_b(&Bs[cur][(wn + ni * 16 + lr) * KC + ks * 32], lane);
#pragma unroll
      for (int mi = 0; mi < 2; ++mi)
#pragma unroll
        for (int ni = 0; ni < 4; ++ni)
          acc[mi][ni] = __builtin_amdgcn_wmma_f32_16x16x32_bf16(
              false, af[mi], false, bfr[ni], (short)0, acc[mi][ni], false, false);
    }
    __builtin_amdgcn_s_wait_tensorcnt(0);
    __syncthreads();
  }

  // epilogue: silu * combine -> bf16 H
  const int rowH = (lane >> 4) * 8;
#pragma unroll
  for (int mi = 0; mi < 2; ++mi)
#pragma unroll
    for (int ni = 0; ni < 4; ++ni)
#pragma unroll
      for (int r = 0; r < 8; ++r) {
        int row = wm + mi * 16 + rowH + r;
        int col = wn + ni * 16 + lr;
        float v = acc[mi][ni][r];
        float s = v / (1.0f + __expf(-v));
        s *= cmb[row];
        H[(long)(mbase + row) * EF + (nbase + col)] = f2bf(s);
      }
}

// ---------------------------------------------------------------------------
// Down GEMM + residual: out[t,d] = zprev[t,d] + sum_k H[t,k] * Bcat[k,d]
// Bcat[k=(e,f), d] = w2bf[e, d, f].  M=2048, N=1024, K=16384.
// ---------------------------------------------------------------------------
__global__ __launch_bounds__(256) void moe_down_kernel(const u16* __restrict__ H,
                                                       const u16* __restrict__ w2bf,
                                                       const float* __restrict__ zprev,
                                                       float* __restrict__ out) {
  const int dbase = blockIdx.x * 128;
  const int mbase = blockIdx.y * 128;
  const int tid = threadIdx.x;
  const int lane = tid & 31, wid = tid >> 5;
  const int wm = (wid & 3) * 32, wn = (wid >> 2) * 64;
  const int lr = lane & 15;

  __shared__ __align__(16) u16 As[2][128 * KC];
  __shared__ __align__(16) u16 Bs[2][128 * KC];

  const u16* gA = H + (long)mbase * EF;
  unsigned ldsA[2] = {(unsigned)(unsigned long long)(void*)&As[0][0],
                      (unsigned)(unsigned long long)(void*)&As[1][0]};
  unsigned ldsB[2] = {(unsigned)(unsigned long long)(void*)&Bs[0][0],
                      (unsigned)(unsigned long long)(void*)&Bs[1][0]};

  if (wid == 0) {
    tdm_load_tile_2d(ldsA[0], gA, 128, KC, EF);
    // k=0 -> expert 0, f=0
    tdm_load_tile_2d(ldsB[0], w2bf + (long)dbase * F_DIM, 128, KC, F_DIM);
  }
  __builtin_amdgcn_s_wait_tensorcnt(0);
  __syncthreads();

  v8f acc[2][4];
#pragma unroll
  for (int mi = 0; mi < 2; ++mi)
#pragma unroll
    for (int ni = 0; ni < 4; ++ni) acc[mi][ni] = vzero8();

  const int KT = EF / KC;  // 256
  for (int kc = 0; kc < KT; ++kc) {
    const int cur = kc & 1, nxt = cur ^ 1;
    if (kc + 1 < KT && wid == 0) {
      int kg = (kc + 1) * KC;
      int ex = kg >> 11;          // kg / F_DIM
      int f = kg & (F_DIM - 1);
      tdm_load_tile_2d(ldsA[nxt], gA + kg, 128, KC, EF);
      tdm_load_tile_2d(ldsB[nxt],
                       w2bf + ((long)(ex * D_DIM + dbase)) * F_DIM + f,
                       128, KC, F_DIM);
    }
#pragma unroll
    for (int ks = 0; ks < 2; ++ks) {
      v16bf af[2], bfr[4];
#pragma unroll
      for (int mi = 0; mi < 2; ++mi)
        af[mi] = load_frag_a(&As[cur][(wm + mi * 16 + lr) * KC + ks * 32], lane);
#pragma unroll
      for (int ni = 0; ni < 4; ++ni)
        bfr[ni] = load_frag_b(&Bs[cur][(wn + ni * 16 + lr) * KC + ks * 32], lane);
#pragma unroll
      for (int mi = 0; mi < 2; ++mi)
#pragma unroll
        for (int ni = 0; ni < 4; ++ni)
          acc[mi][ni] = __builtin_amdgcn_wmma_f32_16x16x32_bf16(
              false, af[mi], false, bfr[ni], (short)0, acc[mi][ni], false, false);
    }
    __builtin_amdgcn_s_wait_tensorcnt(0);
    __syncthreads();
  }

  const int rowH = (lane >> 4) * 8;
#pragma unroll
  for (int mi = 0; mi < 2; ++mi)
#pragma unroll
    for (int ni = 0; ni < 4; ++ni)
#pragma unroll
      for (int r = 0; r < 8; ++r) {
        int row = wm + mi * 16 + rowH + r;
        int col = wn + ni * 16 + lr;
        long idx = (long)(mbase + row) * D_DIM + (dbase + col);
        out[idx] = zprev[idx] + acc[mi][ni][r];
      }
}

// ---------------------------------------------------------------------------
// Host launcher
// ---------------------------------------------------------------------------
extern "C" void kernel_launch(void* const* d_in, const int* in_sizes, int n_in,
                              void* d_out, int out_size, void* d_ws, size_t ws_size,
                              hipStream_t stream) {
  (void)in_sizes; (void)n_in; (void)out_size; (void)ws_size;
  const float* z_in   = (const float*)d_in[0];
  const float* norm_w = (const float*)d_in[1];
  const float* w_mac  = (const float*)d_in[2];
  const float* w_mic  = (const float*)d_in[3];
  const float* w1     = (const float*)d_in[4];
  const float* w2     = (const float*)d_in[5];

  char* ws = (char*)d_ws;
  size_t off = 0;
  auto carve = [&](size_t bytes) -> void* {
    void* p = (void*)(ws + off);
    off = (off + bytes + 255) & ~(size_t)255;
    return p;
  };
  const size_t n_w1 = (size_t)N_EXP * F_DIM * D_DIM;
  const size_t n_w2 = (size_t)N_EXP * D_DIM * F_DIM;
  u16*   w1bf    = (u16*)carve(n_w1 * 2);
  u16*   w2bf    = (u16*)carve(n_w2 * 2);
  u16*   Hbuf    = (u16*)carve((size_t)T_TOK * EF * 2);
  u16*   xn      = (u16*)carve((size_t)T_TOK * D_DIM * 2);
  float* combine = (float*)carve((size_t)T_TOK * N_EXP * 4);
  float* zbuf    = (float*)carve((size_t)T_TOK * D_DIM * 4);

  cvt_bf16_kernel<<<(int)((n_w1 + 255) / 256), 256, 0, stream>>>(w1, w1bf, (int)n_w1);
  cvt_bf16_kernel<<<(int)((n_w2 + 255) / 256), 256, 0, stream>>>(w2, w2bf, (int)n_w2);

  for (int it = 0; it < 2; ++it) {
    const float* zp = (it == 0) ? z_in : zbuf;
    float* zo = (it == 1) ? (float*)d_out : zbuf;
    rmsnorm_kernel<<<T_TOK, 256, 0, stream>>>(zp, norm_w, xn);
    router_kernel<<<T_TOK, 256, 0, stream>>>(xn, w_mac, w_mic, combine);
    moe_up_kernel<<<dim3(EF / 128, T_TOK / 128), 256, 0, stream>>>(xn, w1bf, combine, Hbuf);
    moe_down_kernel<<<dim3(D_DIM / 128, T_TOK / 128), 256, 0, stream>>>(Hbuf, w2bf, zp, zo);
  }
}
